// LSTMAutoregressiveDecoder_31937376813723
// MI455X (gfx1250) — matrix operand order repaired and checked by reading, not compiled, not run on hardware
//
#include <hip/hip_runtime.h>
#include <stdint.h>

// ---------------------------------------------------------------------------
// Problem constants (match reference)
// ---------------------------------------------------------------------------
#define BATCH        2048
#define INPUT_DIM    256
#define DEC_UNITS    1024
#define OUTPUT_STEPS 96
#define OUTPUT_UNITS 256
#define GATES        (4 * DEC_UNITS)   // 4096

// ---------------------------------------------------------------------------
// WMMA fragment types (gfx1250, wave32)
// ---------------------------------------------------------------------------
typedef __bf16 v16bf __attribute__((ext_vector_type(16)));
typedef __bf16 v8bf  __attribute__((ext_vector_type(8)));
typedef float  v8f   __attribute__((ext_vector_type(8)));

// round-to-nearest-even f32 -> bf16 bit pattern
__device__ __forceinline__ uint16_t f32_to_bf16_bits(float f) {
    union { float f; uint32_t u; } v; v.f = f;
    uint32_t r = v.u + 0x7FFFu + ((v.u >> 16) & 1u);
    return (uint16_t)(r >> 16);
}

// A fragment: 16x32 bf16 tile, row-major source (row stride lda elements).
// ISA layout: lanes 0-15 hold M=lane: VGPR0-3 = K 0..7, VGPR4-7 = K 16..23;
// lanes 16-31 hold M=lane-15? -> M=lane&15 with K 8..15 / 24..31.
// base must point at (row0, k0) of the tile.
__device__ __forceinline__ v16bf load_a_frag(const __bf16* base, int lda, int lane) {
    const int m    = lane & 15;
    const int half = lane >> 4;
    const __bf16* p = base + (size_t)m * lda + half * 8;
    union { v16bf v; v8bf h[2]; } u;
    u.h[0] = *(const v8bf*)(p);        // K: half*8 .. +8
    u.h[1] = *(const v8bf*)(p + 16);   // K: 16 + half*8 .. +8
    return u.v;
}

// B fragment: 32x16 bf16 tile taken from a pre-TRANSPOSED weight matrix
// WT[N_total][K_total] (row-major, row stride ldk). Lane holds column
// n = lane&15, K (lane>>4)*8 .. +8 and 16 + (lane>>4)*8 .. +8 (same as A).
// baseT must point at (col0-row of WT, k0).
__device__ __forceinline__ v16bf load_b_frag(const __bf16* baseT, int ldk, int lane) {
    const int n    = lane & 15;
    const int half = lane >> 4;
    const __bf16* p = baseT + (size_t)n * ldk + half * 8;
    union { v16bf v; v8bf h[2]; } u;
    u.h[0] = *(const v8bf*)(p);
    u.h[1] = *(const v8bf*)(p + 16);
    return u.v;
}

__device__ __forceinline__ float sigmoidf_fast(float x) {
    return 1.0f / (1.0f + __expf(-x));
}

// ---------------------------------------------------------------------------
// Prologue: transpose-convert fp32 W[K][N] -> bf16 WT[N][K]
// block (32,8), grid (N/32, K/32). K,N are multiples of 32 here.
// ---------------------------------------------------------------------------
__global__ __launch_bounds__(256) void transpose_to_bf16_kernel(
    const float* __restrict__ W, uint16_t* __restrict__ WT, int K, int N) {
    __shared__ float tile[32][33];
    const int n0 = blockIdx.x * 32;
    const int k0 = blockIdx.y * 32;
    const int tx = threadIdx.x, ty = threadIdx.y;
#pragma unroll
    for (int i = 0; i < 32; i += 8) {
        tile[ty + i][tx] = W[(size_t)(k0 + ty + i) * N + (n0 + tx)];
    }
    __syncthreads();
#pragma unroll
    for (int i = 0; i < 32; i += 8) {
        WT[(size_t)(n0 + ty + i) * K + (k0 + tx)] = f32_to_bf16_bits(tile[tx][ty + i]);
    }
}

__global__ __launch_bounds__(256) void f32_to_bf16_kernel(
    const float* __restrict__ src, uint16_t* __restrict__ dst, int n) {
    int i = blockIdx.x * 256 + threadIdx.x;
    if (i < n) dst[i] = f32_to_bf16_bits(src[i]);
}

// ---------------------------------------------------------------------------
// Fused gates GEMM + LSTM pointwise.
//   z = x@Wk + h@Wr + b   (K = 256 + 1024, N = 4096 as 4 gates x 1024)
//   c_new = sigm(zf)*c + sigm(zi)*tanh(zg);  h_new = sigm(zo)*tanh(c_new)
// Block: 256 threads = 8 waves, arranged 2(M) x 4(N); per-wave tile:
// 32(M) x 16(N) x 4 gates  -> 8 f32 accumulator fragments.
// Grid: (BATCH/64, DEC_UNITS/64) = (32, 16).
// ---------------------------------------------------------------------------
__global__ __launch_bounds__(256) void lstm_gates_kernel(
    const __bf16*  __restrict__ x,     // [BATCH, INPUT_DIM] bf16
    const __bf16*  __restrict__ hprev, // [BATCH, DEC_UNITS] bf16
    const __bf16*  __restrict__ WkT,   // [GATES, INPUT_DIM] bf16 (transposed)
    const __bf16*  __restrict__ WrT,   // [GATES, DEC_UNITS] bf16 (transposed)
    const float*   __restrict__ bias,  // [GATES] f32
    float*         __restrict__ c,     // [BATCH, DEC_UNITS] f32 (in/out)
    uint16_t*      __restrict__ hnext) // [BATCH, DEC_UNITS] bf16
{
    const int lane = threadIdx.x & 31;
    const int wave = threadIdx.x >> 5;
    const int wn   = wave & 3;          // 0..3 along N
    const int wm   = wave >> 2;         // 0..1 along M
    const int row0 = blockIdx.x * 64 + wm * 32;
    const int col0 = blockIdx.y * 64 + wn * 16;
    const int ncol = lane & 15;

    v8f acc[2][4];
#pragma unroll
    for (int g = 0; g < 4; ++g) {
        const float bv = bias[g * DEC_UNITS + col0 + ncol];
#pragma unroll
        for (int mi = 0; mi < 2; ++mi)
#pragma unroll
            for (int r = 0; r < 8; ++r) acc[mi][g][r] = bv;
    }

    // ---- x @ Wk : K = INPUT_DIM ----
    for (int k0 = 0; k0 < INPUT_DIM; k0 += 32) {
        v16bf a0 = load_a_frag(x + (size_t)(row0     ) * INPUT_DIM + k0, INPUT_DIM, lane);
        v16bf a1 = load_a_frag(x + (size_t)(row0 + 16) * INPUT_DIM + k0, INPUT_DIM, lane);
#pragma unroll
        for (int g = 0; g < 4; ++g) {
            v16bf bf = load_b_frag(WkT + (size_t)(g * DEC_UNITS + col0) * INPUT_DIM + k0,
                                   INPUT_DIM, lane);
            acc[0][g] = __builtin_amdgcn_wmma_f32_16x16x32_bf16(
                false, a0, false, bf, (short)0, acc[0][g], false, false);
            acc[1][g] = __builtin_amdgcn_wmma_f32_16x16x32_bf16(
                false, a1, false, bf, (short)0, acc[1][g], false, false);
        }
    }
    // ---- h @ Wr : K = DEC_UNITS ----
    for (int k0 = 0; k0 < DEC_UNITS; k0 += 32) {
        v16bf a0 = load_a_frag(hprev + (size_t)(row0     ) * DEC_UNITS + k0, DEC_UNITS, lane);
        v16bf a1 = load_a_frag(hprev + (size_t)(row0 + 16) * DEC_UNITS + k0, DEC_UNITS, lane);
#pragma unroll
        for (int g = 0; g < 4; ++g) {
            v16bf bf = load_b_frag(WrT + (size_t)(g * DEC_UNITS + col0) * DEC_UNITS + k0,
                                   DEC_UNITS, lane);
            acc[0][g] = __builtin_amdgcn_wmma_f32_16x16x32_bf16(
                false, a0, false, bf, (short)0, acc[0][g], false, false);
            acc[1][g] = __builtin_amdgcn_wmma_f32_16x16x32_bf16(
                false, a1, false, bf, (short)0, acc[1][g], false, false);
        }
    }

    // ---- fused LSTM pointwise epilogue ----
    // f32 C layout: lane holds column ncol; rows (lane>>4)*8 + r per VGPR r.
    const int rbase = (lane >> 4) * 8;
#pragma unroll
    for (int mi = 0; mi < 2; ++mi) {
#pragma unroll
        for (int r = 0; r < 8; ++r) {
            const int row = row0 + mi * 16 + rbase + r;
            const size_t idx = (size_t)row * DEC_UNITS + col0 + ncol;
            const float zi = acc[mi][0][r];
            const float zf = acc[mi][1][r];
            const float zg = acc[mi][2][r];
            const float zo = acc[mi][3][r];
            const float ig = sigmoidf_fast(zi);
            const float fg = sigmoidf_fast(zf);
            const float og = sigmoidf_fast(zo);
            const float cn = fg * c[idx] + ig * tanhf(zg);
            c[idx] = cn;
            hnext[idx] = f32_to_bf16_bits(og * tanhf(cn));
        }
    }
}

// ---------------------------------------------------------------------------
// Output projection: pred = h @ Wd + bd. Writes fp32 output slice for step t
// and bf16 next-step input x. Same wave tiling: 2(M) x 4(N) waves,
// per-wave 32x16. Grid: (BATCH/64, OUTPUT_UNITS/64) = (32, 4).
// ---------------------------------------------------------------------------
__global__ __launch_bounds__(256) void dense_out_kernel(
    const __bf16* __restrict__ h,     // [BATCH, DEC_UNITS] bf16
    const __bf16* __restrict__ WdT,   // [OUTPUT_UNITS, DEC_UNITS] bf16 (transposed)
    const float*  __restrict__ bd,    // [OUTPUT_UNITS] f32
    float*        __restrict__ out,   // [BATCH, OUTPUT_STEPS, OUTPUT_UNITS] f32
    uint16_t*     __restrict__ xnext, // [BATCH, INPUT_DIM] bf16
    int t)
{
    const int lane = threadIdx.x & 31;
    const int wave = threadIdx.x >> 5;
    const int wn   = wave & 3;
    const int wm   = wave >> 2;
    const int row0 = blockIdx.x * 64 + wm * 32;
    const int col0 = blockIdx.y * 64 + wn * 16;
    const int ncol = lane & 15;

    v8f acc[2];
    {
        const float bv = bd[col0 + ncol];
#pragma unroll
        for (int mi = 0; mi < 2; ++mi)
#pragma unroll
            for (int r = 0; r < 8; ++r) acc[mi][r] = bv;
    }

    for (int k0 = 0; k0 < DEC_UNITS; k0 += 32) {
        v16bf a0 = load_a_frag(h + (size_t)(row0     ) * DEC_UNITS + k0, DEC_UNITS, lane);
        v16bf a1 = load_a_frag(h + (size_t)(row0 + 16) * DEC_UNITS + k0, DEC_UNITS, lane);
        v16bf bf = load_b_frag(WdT + (size_t)col0 * DEC_UNITS + k0, DEC_UNITS, lane);
        acc[0] = __builtin_amdgcn_wmma_f32_16x16x32_bf16(
            false, a0, false, bf, (short)0, acc[0], false, false);
        acc[1] = __builtin_amdgcn_wmma_f32_16x16x32_bf16(
            false, a1, false, bf, (short)0, acc[1], false, false);
    }

    const int rbase = (lane >> 4) * 8;
#pragma unroll
    for (int mi = 0; mi < 2; ++mi) {
#pragma unroll
        for (int r = 0; r < 8; ++r) {
            const int row = row0 + mi * 16 + rbase + r;
            const int col = col0 + ncol;
            const float p = acc[mi][r];
            out[(size_t)row * (OUTPUT_STEPS * OUTPUT_UNITS) + (size_t)t * OUTPUT_UNITS + col] = p;
            xnext[(size_t)row * INPUT_DIM + col] = f32_to_bf16_bits(p);
        }
    }
}

// ---------------------------------------------------------------------------
// Host-side orchestration (graph-capture safe: only kernel launches + one
// device-to-device hipMemcpyAsync on `stream`).
// ---------------------------------------------------------------------------
extern "C" void kernel_launch(void* const* d_in, const int* in_sizes, int n_in,
                              void* d_out, int out_size, void* d_ws, size_t ws_size,
                              hipStream_t stream) {
    (void)in_sizes; (void)n_in; (void)out_size; (void)ws_size;

    const float* last_input = (const float*)d_in[0];
    const float* h0         = (const float*)d_in[1];
    const float* c0         = (const float*)d_in[2];
    const float* Wk         = (const float*)d_in[3];
    const float* Wr         = (const float*)d_in[4];
    const float* b          = (const float*)d_in[5];
    const float* Wd         = (const float*)d_in[6];
    const float* bd         = (const float*)d_in[7];
    float* out = (float*)d_out;

    // workspace carve (256B aligned)
    char* w = (char*)d_ws;
    auto carve = [&](size_t bytes) {
        char* p = w;
        w += (bytes + 255) & ~(size_t)255;
        return p;
    };
    uint16_t* WkT  = (uint16_t*)carve((size_t)GATES * INPUT_DIM * 2);        // 2 MB
    uint16_t* WrT  = (uint16_t*)carve((size_t)GATES * DEC_UNITS * 2);        // 8 MB
    uint16_t* WdT  = (uint16_t*)carve((size_t)OUTPUT_UNITS * DEC_UNITS * 2); // 0.5 MB
    uint16_t* x_bf = (uint16_t*)carve((size_t)BATCH * INPUT_DIM * 2);        // 1 MB
    uint16_t* h_bf0 = (uint16_t*)carve((size_t)BATCH * DEC_UNITS * 2);       // 4 MB
    uint16_t* h_bf1 = (uint16_t*)carve((size_t)BATCH * DEC_UNITS * 2);       // 4 MB
    float*    c_ws  = (float*)   carve((size_t)BATCH * DEC_UNITS * 4);       // 8 MB

    // --- prologue: weight transpose-convert + state init ---
    dim3 tb(32, 8);
    transpose_to_bf16_kernel<<<dim3(GATES / 32, INPUT_DIM / 32), tb, 0, stream>>>(
        Wk, WkT, INPUT_DIM, GATES);
    transpose_to_bf16_kernel<<<dim3(GATES / 32, DEC_UNITS / 32), tb, 0, stream>>>(
        Wr, WrT, DEC_UNITS, GATES);
    transpose_to_bf16_kernel<<<dim3(OUTPUT_UNITS / 32, DEC_UNITS / 32), tb, 0, stream>>>(
        Wd, WdT, DEC_UNITS, OUTPUT_UNITS);

    {
        int n = BATCH * INPUT_DIM;
        f32_to_bf16_kernel<<<(n + 255) / 256, 256, 0, stream>>>(last_input, x_bf, n);
    }
    {
        int n = BATCH * DEC_UNITS;
        f32_to_bf16_kernel<<<(n + 255) / 256, 256, 0, stream>>>(h0, h_bf0, n);
    }
    hipMemcpyAsync(c_ws, c0, (size_t)BATCH * DEC_UNITS * 4,
                   hipMemcpyDeviceToDevice, stream);

    // --- 96 autoregressive steps ---
    const dim3 gates_grid(BATCH / 64, DEC_UNITS / 64);   // (32, 16)
    const dim3 dense_grid(BATCH / 64, OUTPUT_UNITS / 64); // (32, 4)
    for (int t = 0; t < OUTPUT_STEPS; ++t) {
        uint16_t* h_prev = (t & 1) ? h_bf1 : h_bf0;
        uint16_t* h_next = (t & 1) ? h_bf0 : h_bf1;

        lstm_gates_kernel<<<gates_grid, 256, 0, stream>>>(
            (const __bf16*)x_bf, (const __bf16*)h_prev,
            (const __bf16*)WkT, (const __bf16*)WrT,
            b, c_ws, h_next);

        dense_out_kernel<<<dense_grid, 256, 0, stream>>>(
            (const __bf16*)h_next, (const __bf16*)WdT,
            bd, out, x_bf, t);
    }
}
